// AutocorrF0Extractor_23948737642966
// MI455X (gfx1250) — matrix operand order repaired and checked by reading, not compiled, not run on hardware
//
#include <hip/hip_runtime.h>
#include <math.h>

typedef __attribute__((ext_vector_type(16))) _Float16 v16h;
typedef __attribute__((ext_vector_type(8)))  _Float16 v8h;
typedef __attribute__((ext_vector_type(8)))  float    v8f;

#define FRAME   1024
#define HOP     256
#define LAG_MIN 44          // int(22050/500)
#define NREG    324         // lags 44..367 inclusive
#define PAD     256         // >= 16*15 = 240 negative reach of A rows
#define WTOT    1856        // PAD + 1600 (covers max B read index 1594)
#define NCHUNK  40          // 40*32 = 1280 >= 1264 reduction coverage

// ---------------------------------------------------------------------------
// Kernel 1: one frame per 64-thread block (2 wave32's).
//   wave 0 -> ACF tile lags [44..299], wave 1 -> tile lags [300..555]
//   via V_WMMA_F32_16X16X32_F16 over a Toeplitz factorization of the ACF.
// ---------------------------------------------------------------------------
__global__ __launch_bounds__(64) void f0_frame_kernel(
    const float* __restrict__ wav, float* __restrict__ out,
    float* __restrict__ ws, int n_frames, int T)
{
  __shared__ __align__(16) _Float16 wpad[WTOT];  // zero-padded windowed frame (f16)
  __shared__ float fr[FRAME];                    // raw frame (f32) for energy/zcr
  __shared__ float acfbuf[512];                  // unnormalized ACF, lag-44 relative
  __shared__ float redE[64], redR[64], redZ[64];
  __shared__ float bestv[64];
  __shared__ int   besti[64];

  const int tid = threadIdx.x;
  const int f   = blockIdx.x;
  const long base = (long)f * HOP;

  // ---- phase 1: zero pads, load+window frame, energy & r0 partials ----
  for (int i = tid; i < WTOT; i += 64) wpad[i] = (_Float16)0.0f;

  float pE = 0.0f, pR = 0.0f;
  for (int t = tid; t < FRAME; t += 64) {       // same threads zero & fill each slot
    long gi = base + t;
    float x = (gi < (long)T) ? wav[gi] : 0.0f;
    fr[t] = x;
    float dt  = ((float)t - 512.0f) * (1.0f / 256.0f);
    float win = __expf(-0.5f * dt * dt);
    float w   = x * win;
    wpad[PAD + t] = (_Float16)w;
    pE += x * x;
    pR += w * w;
  }
  redE[tid] = pE; redR[tid] = pR;
  __syncthreads();

  // ---- zcr partial (fr fully resident) ----
  float pZ = 0.0f;
  for (int t = tid; t < FRAME; t += 64)
    if (t >= 1) pZ += (fr[t] * fr[t - 1] < 0.0f) ? 1.0f : 0.0f;
  redZ[tid] = pZ;

  // ---- phase 2: WMMA autocorrelation, one 256-lag tile per wave ----
  const int lane = tid & 31;
  const int M  = lane & 15;        // A row / B column / C column index
  const int hb = lane >> 4;        // half-K selector
  const int l0 = (tid < 32) ? LAG_MIN : (LAG_MIN + 256);   // wave-uniform

  v8f acc = {};
  const int iA0 = PAD + 8 * hb - 16 * M;        // 16-byte aligned (all terms %8==0)
  const int iB0 = PAD + l0 + 8 * hb + M;        // 2-byte aligned (per-lane shift)
  for (int c = 0; c < NCHUNK; ++c) {
    const int s = c * 32;
    const v8h alo = *(const v8h*)&wpad[iA0 + s];        // K = 8hb..8hb+7
    const v8h ahi = *(const v8h*)&wpad[iA0 + s + 16];   // K = 16+8hb..16+8hb+7
    v16h a, b;
#pragma unroll
    for (int e = 0; e < 8; ++e) { a[e] = alo[e]; a[e + 8] = ahi[e]; }
#pragma unroll
    for (int e = 0; e < 16; ++e) b[e] = wpad[iB0 + s + e + (e & 8)];
    acc = __builtin_amdgcn_wmma_f32_16x16x32_f16(
        false, a, false, b, (short)0, acc, false, false);
  }
#pragma unroll
  for (int r = 0; r < 8; ++r) {
    int lagrel = (l0 - LAG_MIN) + 16 * (r + 8 * hb) + M;  // C[M',N] -> lag
    acfbuf[lagrel] = acc[r];
  }
  __syncthreads();

  // ---- phase 3: argmax over region (scale-invariant on unnormalized ACF) ----
  float bv = -3.0e38f; int bi = 0;
  for (int r = tid; r < NREG; r += 64) {
    float v = acfbuf[r];
    if (v > bv || (v == bv && r < bi)) { bv = v; bi = r; }
  }
  bestv[tid] = bv; besti[tid] = bi;
  __syncthreads();

  if (tid == 0) {
    float sE = 0.f, sR = 0.f, sZ = 0.f;
    for (int i = 0; i < 64; ++i) { sE += redE[i]; sR += redR[i]; sZ += redZ[i]; }
    float energy = sqrtf(sE * (1.0f / FRAME));
    float zcr    = sZ * (1.0f / (FRAME - 1));
    float inv_r0 = 1.0f / fmaxf(sR, 1e-8f);

    float v = bestv[0]; int pk = besti[0];
    for (int i = 1; i < 64; ++i) {
      float vi = bestv[i]; int ii = besti[i];
      if (vi > v || (vi == v && ii < pk)) { v = vi; pk = ii; }
    }
    float strength = v * inv_r0;
    int pk_c = pk < 1 ? 1 : (pk > NREG - 2 ? NREG - 2 : pk);
    float y0 = acfbuf[pk_c - 1] * inv_r0;
    float y1 = acfbuf[pk_c]     * inv_r0;
    float y2 = acfbuf[pk_c + 1] * inv_r0;
    float denom = 2.0f * (2.0f * y1 - y0 - y2);
    float delta = (fabsf(denom) > 1e-8f) ? (y0 - y2) / denom : 0.0f;
    if (!(pk > 0 && pk < NREG - 1)) delta = 0.0f;
    float lag = (float)LAG_MIN + (float)pk + delta;
    float f0c = 22050.0f / lag;

    out[n_frames + f]        = energy;     // energy output slot
    ws[1 + f]                = f0c;        // pre-mask f0
    ws[1 + n_frames + f]     = strength;
    ws[1 + 2 * n_frames + f] = zcr;
    // global max energy: energy >= 0, so uint bit pattern is order-preserving
    atomicMax((unsigned int*)ws, __float_as_uint(energy));
  }
}

// ---------------------------------------------------------------------------
// Kernel 2: apply global energy threshold + voicing mask.
// ---------------------------------------------------------------------------
__global__ void f0_mask_kernel(float* __restrict__ out,
                               const float* __restrict__ ws, int n)
{
  int f = blockIdx.x * blockDim.x + threadIdx.x;
  if (f >= n) return;
  float maxE = __uint_as_float(((const unsigned int*)ws)[0]);
  float thr  = fmaxf(maxE, 1e-8f) * 0.05f;
  float energy   = out[n + f];
  float f0c      = ws[1 + f];
  float strength = ws[1 + n + f];
  float zcr      = ws[1 + 2 * n + f];
  bool m = (strength >= 0.45f) && (energy > thr) && (zcr < 0.3f);
  out[f]         = m ? f0c : 0.0f;
  out[2 * n + f] = m ? 1.0f : 0.0f;
}

extern "C" void kernel_launch(void* const* d_in, const int* in_sizes, int n_in,
                              void* d_out, int out_size, void* d_ws, size_t ws_size,
                              hipStream_t stream) {
  const float* wav = (const float*)d_in[0];
  const int T = in_sizes[0];
  int n = (T - FRAME) / HOP + 1;
  if (n < 1) n = 1;
  float* out = (float*)d_out;
  float* ws  = (float*)d_ws;

  hipMemsetAsync(d_ws, 0, sizeof(unsigned int), stream);   // init atomic max slot
  f0_frame_kernel<<<n, 64, 0, stream>>>(wav, out, ws, n, T);
  const int thr = 256;
  f0_mask_kernel<<<(n + thr - 1) / thr, thr, 0, stream>>>(out, ws, n);
}
